// TinyBioBERT_86509231276739
// MI455X (gfx1250) — matrix operand order repaired and checked by reading, not compiled, not run on hardware
//
#include <hip/hip_runtime.h>
#include <hip/hip_bf16.h>
#include <math.h>

// Problem constants (match reference)
#define B_    16
#define S_    512
#define D_    1024
#define H_    16
#define L_    8
#define FF_   4096
#define DH_   64
#define MROWS (B_ * S_)        // 8192
#define LNEPS 1e-12f

typedef __attribute__((ext_vector_type(16))) __bf16 v16bf;
typedef __attribute__((ext_vector_type(8)))  __bf16 v8bf;
typedef __attribute__((ext_vector_type(4)))  __bf16 v4bf;
typedef __attribute__((ext_vector_type(8)))  float  v8f;

// Native f32 -> bf16 (RNE); backend emits v_cvt_*bf16* on gfx1250
static __device__ __forceinline__ __bf16 f2bf(float f) { return (__bf16)f; }

static __device__ __forceinline__ v4bf pack4(float4 f) {
    v4bf r;
    r[0] = f2bf(f.x); r[1] = f2bf(f.y); r[2] = f2bf(f.z); r[3] = f2bf(f.w);
    return r;
}
// Load one lane's 16-bf16 fragment (32 contiguous bytes -> 2x ds_load_b128)
static __device__ __forceinline__ v16bf ld_frag(const __bf16* p) {
    v8bf lo = *(const v8bf*)(p);
    v8bf hi = *(const v8bf*)(p + 8);
    return __builtin_shufflevector(lo, hi, 0, 1, 2, 3, 4, 5, 6, 7,
                                   8, 9, 10, 11, 12, 13, 14, 15);
}
// A-fragment element index for K-offset kk (0..31):  lane-half h=(kk>>3)&1,
// element e = (kk&7) + 8*((kk>>4)&1)      (inverse of ISA 16-bit A layout)
// B-fragment: same mapping as A (lane-half h=(kk>>3)&1, element e as above)

// ---------------------------------------------------------------------------
// Embeddings + LayerNorm (one block per row)
// ---------------------------------------------------------------------------
__global__ void __launch_bounds__(256)
embed_ln_kernel(const int* __restrict__ ids, const int* __restrict__ tts,
                const float* __restrict__ tok, const float* __restrict__ pos,
                const float* __restrict__ typ, const float* __restrict__ g,
                const float* __restrict__ b, float* __restrict__ out) {
    const int row = blockIdx.x;
    const int s   = row % S_;
    const int id  = ids[row];
    const int tt  = tts[row];
    const float* pt = tok + (size_t)id * D_;
    const float* pp = pos + (size_t)s  * D_;
    const float* py = typ + (size_t)tt * D_;
    __shared__ float red[256], red2[256];
    const int t = threadIdx.x;
    float vals[4];
    float s1 = 0.f, s2 = 0.f;
    #pragma unroll
    for (int i = 0; i < 4; ++i) {
        int c = t + 256 * i;
        float v = pt[c] + pp[c] + py[c];
        vals[i] = v; s1 += v; s2 += v * v;
    }
    red[t] = s1; red2[t] = s2;
    __syncthreads();
    for (int off = 128; off > 0; off >>= 1) {
        if (t < off) { red[t] += red[t + off]; red2[t] += red2[t + off]; }
        __syncthreads();
    }
    const float mean = red[0] * (1.0f / D_);
    const float var  = red2[0] * (1.0f / D_) - mean * mean;
    const float inv  = rsqrtf(var + LNEPS);
    #pragma unroll
    for (int i = 0; i < 4; ++i) {
        int c = t + 256 * i;
        out[(size_t)row * D_ + c] = (vals[i] - mean) * inv * g[c] + b[c];
    }
}

// ---------------------------------------------------------------------------
// LayerNorm (one block per row)
// ---------------------------------------------------------------------------
__global__ void __launch_bounds__(256)
ln_kernel(const float* __restrict__ in, const float* __restrict__ g,
          const float* __restrict__ b, float* __restrict__ out) {
    const int row = blockIdx.x;
    const float* p = in + (size_t)row * D_;
    __shared__ float red[256], red2[256];
    const int t = threadIdx.x;
    float vals[4];
    float s1 = 0.f, s2 = 0.f;
    #pragma unroll
    for (int i = 0; i < 4; ++i) {
        float v = p[t + 256 * i];
        vals[i] = v; s1 += v; s2 += v * v;
    }
    red[t] = s1; red2[t] = s2;
    __syncthreads();
    for (int off = 128; off > 0; off >>= 1) {
        if (t < off) { red[t] += red[t + off]; red2[t] += red2[t + off]; }
        __syncthreads();
    }
    const float mean = red[0] * (1.0f / D_);
    const float var  = red2[0] * (1.0f / D_) - mean * mean;
    const float inv  = rsqrtf(var + LNEPS);
    #pragma unroll
    for (int i = 0; i < 4; ++i) {
        int c = t + 256 * i;
        out[(size_t)row * D_ + c] = (vals[i] - mean) * inv * g[c] + b[c];
    }
}

// ---------------------------------------------------------------------------
// bf16-WMMA GEMM:  out = [gelu](A @ W + bias) [+ res]
// Block tile 128(M) x 128(N), BK = 32, double-buffered LDS software pipeline:
// next tile's global_load_b128s are issued before the WMMA block, one barrier
// per K-step. 8 waves: 4 along M x 2 along N; each wave owns 2x4 WMMA tiles.
// LDS tiles are stored in WMMA fragment order (per-lane 32B contiguous).
// ---------------------------------------------------------------------------
template <int GELU, int RES>
__global__ void __launch_bounds__(256)
gemm_ws(const float* __restrict__ A, const float* __restrict__ W,
        const float* __restrict__ bias, const float* __restrict__ res,
        float* __restrict__ out, int M, int N, int K) {
    const int m0 = blockIdx.x * 128;
    const int n0 = blockIdx.y * 128;
    // [tile][lane][fragment elem], inner padded 16->24 (48B stride, 16B aligned)
    __shared__ __attribute__((aligned(16))) __bf16 AF0[8][32][24], AF1[8][32][24];
    __shared__ __attribute__((aligned(16))) __bf16 BF0[8][32][24], BF1[8][32][24];
    const int tid  = threadIdx.x;
    const int lane = tid & 31;
    const int w    = tid >> 5;
    const int wm   = (w & 3) * 32;   // 4 waves along M
    const int wn   = (w >> 2) * 64;  // 2 waves along N
    const int half = lane >> 4;
    const int l16  = lane & 15;

    v8f acc[8] = {};                 // acc[tm*4 + tn]
    float4 ra[4], rb[4];

    // issue global loads for K-step kk into registers
    auto load_regs = [&](int kk) {
        #pragma unroll
        for (int i = 0; i < 4; ++i) {
            const int idx = tid + 256 * i;
            ra[i] = *(const float4*)(A + (size_t)(m0 + (idx >> 3)) * K + kk + (idx & 7) * 4);
        }
        #pragma unroll
        for (int i = 0; i < 4; ++i) {
            const int idx = tid + 256 * i;
            rb[i] = *(const float4*)(W + (size_t)(kk + (idx >> 5)) * N + n0 + (idx & 31) * 4);
        }
    };
    // convert + scatter registers into fragment-ordered LDS buffers
    auto store_stage = [&](__bf16 (*AFb)[32][24], __bf16 (*BFb)[32][24]) {
        #pragma unroll
        for (int i = 0; i < 4; ++i) {
            const int idx = tid + 256 * i;
            const int r = idx >> 3;               // row 0..127
            const int k = (idx & 7) * 4;          // k 0,4,...,28
            const int h  = (k >> 3) & 1;
            const int e0 = (k & 7) + 8 * (k >> 4);
            *(v4bf*)&AFb[r >> 4][h * 16 + (r & 15)][e0] = pack4(ra[i]);
        }
        #pragma unroll
        for (int i = 0; i < 4; ++i) {
            const int idx = tid + 256 * i;
            const int r = idx >> 5;               // k-row 0..31
            const int c = (idx & 31) * 4;         // col 0,4,...,124
            const v4bf p = pack4(rb[i]);
            const int nt = c >> 4, c0 = c & 15;
            const int h = (r >> 3) & 1, j = (r & 7) + 8 * (r >> 4);
            BFb[nt][h * 16 + c0 + 0][j] = p[0];
            BFb[nt][h * 16 + c0 + 1][j] = p[1];
            BFb[nt][h * 16 + c0 + 2][j] = p[2];
            BFb[nt][h * 16 + c0 + 3][j] = p[3];
        }
    };
    // fragment loads + 8 WMMAs from one buffer pair
    auto compute = [&](const __bf16 (*AFb)[32][24], const __bf16 (*BFb)[32][24]) {
        v16bf af[2], bfr[4];
        #pragma unroll
        for (int tm = 0; tm < 2; ++tm)
            af[tm] = ld_frag(&AFb[(w & 3) * 2 + tm][lane][0]);
        #pragma unroll
        for (int tn = 0; tn < 4; ++tn)
            bfr[tn] = ld_frag(&BFb[(w >> 2) * 4 + tn][lane][0]);
        #pragma unroll
        for (int tm = 0; tm < 2; ++tm)
            #pragma unroll
            for (int tn = 0; tn < 4; ++tn)
                acc[tm * 4 + tn] = __builtin_amdgcn_wmma_f32_16x16x32_bf16(
                    false, af[tm], false, bfr[tn], (short)0, acc[tm * 4 + tn],
                    false, false);
    };

    load_regs(0);
    store_stage(AF0, BF0);
    for (int k0 = 0; k0 < K; k0 += 64) {         // K % 64 == 0 here
        // phase 0: compute buf0 (k0), stage buf1 (k0+32, always exists)
        load_regs(k0 + 32);
        __syncthreads();
        compute(AF0, BF0);
        store_stage(AF1, BF1);
        // phase 1: compute buf1 (k0+32), stage buf0 (k0+64, if any)
        if (k0 + 64 < K) load_regs(k0 + 64);
        __syncthreads();
        compute(AF1, BF1);
        if (k0 + 64 < K) store_stage(AF0, BF0);
    }

    // Epilogue: bias (+ gelu) (+ residual)
    #pragma unroll
    for (int tm = 0; tm < 2; ++tm) {
        #pragma unroll
        for (int tn = 0; tn < 4; ++tn) {
            const int colg = n0 + wn + tn * 16 + l16;
            const float bval = bias[colg];
            #pragma unroll
            for (int r = 0; r < 8; ++r) {
                const int rowg = m0 + wm + tm * 16 + 8 * half + r;
                float v = acc[tm * 4 + tn][r] + bval;
                if (GELU) v = 0.5f * v * (1.0f + erff(v * 0.70710678118654752f));
                const size_t o = (size_t)rowg * N + colg;
                if (RES) v += res[o];
                out[o] = v;
            }
        }
    }
}

// ---------------------------------------------------------------------------
// Flash attention: one block per (q-tile of 64, b*h). Tiles staged to LDS in
// WMMA fragment order; Q fragments hoisted to registers (loop-invariant).
// ---------------------------------------------------------------------------
__global__ void __launch_bounds__(256)
attn_kernel(const float* __restrict__ Q, const float* __restrict__ Km,
            const float* __restrict__ V, float* __restrict__ ctx) {
    const int qt = blockIdx.x;            // 0..7
    const int bh = blockIdx.y;            // 0..255
    const int b  = bh / H_;
    const int h  = bh % H_;
    const int qbase = qt * 64;
    const size_t base = (size_t)b * S_ * D_ + (size_t)h * DH_;

    // Fragment-ordered tiles. QF/PF: A-layout; KF/VF: B-layout.
    __shared__ __attribute__((aligned(16))) __bf16 QF[4][2][32][24]; // [qtile][dh-kstep]
    __shared__ __attribute__((aligned(16))) __bf16 KF[2][4][32][24]; // [dh-kstep][keytile]
    __shared__ __attribute__((aligned(16))) __bf16 VF[2][4][32][24]; // [key-kstep][dhtile]
    __shared__ __attribute__((aligned(16))) __bf16 PF[4][2][32][16]; // [qtile][key-kstep]
    __shared__ float Ss[64][65];
    __shared__ float m_i[64], l_i[64], corr[64];

    const int tid  = threadIdx.x;
    const int lane = tid & 31;
    const int w    = tid >> 5;
    const int half = lane >> 4;
    const int l16  = lane & 15;
    const int mt   = w & 3;               // query tile (16 rows) per wave
    const int rowg = mt * 16;
    const int colg = (w >> 2) * 32;       // 2 waves along columns (keys / dh)

    // ---- stage Q (64x64) in A-fragment order ----
    #pragma unroll
    for (int i = 0; i < 4; ++i) {
        const int idx = tid + 256 * i;            // 0..1023 float4's
        const int r = idx >> 4;                   // query row 0..63
        const int c = (idx & 15) * 4;             // dh 0,4,...,60
        const float4 t4 = *(const float4*)(Q + base + (size_t)(qbase + r) * D_ + c);
        const int kk = c & 31;
        const int hh = (kk >> 3) & 1;
        const int e0 = (kk & 7) + 8 * (kk >> 4);
        *(v4bf*)&QF[r >> 4][c >> 5][hh * 16 + (r & 15)][e0] = pack4(t4);
    }
    if (tid < 64) { m_i[tid] = -1e30f; l_i[tid] = 0.f; }
    __syncthreads();

    // Hoist this wave's Q fragments (invariant over key tiles)
    v16bf aq[2];
    #pragma unroll
    for (int ks = 0; ks < 2; ++ks)
        aq[ks] = ld_frag(&QF[mt][ks][lane][0]);

    v8f accO[2] = {};
    const float scale = 0.125f;           // BETA / sqrt(DH)

    for (int kt = 0; kt < 8; ++kt) {
        __syncthreads();                  // prev-iter consumers done
        const int kbase = kt * 64;
        // ---- stage K tile: B-layout over (k=dh, n=key) ----
        #pragma unroll
        for (int i = 0; i < 4; ++i) {
            const int idx = tid + 256 * i;
            const int r = idx >> 4;               // key 0..63
            const int c = (idx & 15) * 4;         // dh
            const float4 t4 = *(const float4*)(Km + base + (size_t)(kbase + r) * D_ + c);
            // lane = ((dh>>3)&1)*16 + (key&15), elem = (dh&7) + 8*((dh>>4)&1) (4 consecutive)
            *(v4bf*)&KF[c >> 5][r >> 4][((c >> 3) & 1) * 16 + (r & 15)][(c & 7) + 8 * ((c >> 4) & 1)] = pack4(t4);
        }
        // ---- stage V tile: B-layout over (k=key, n=dh) ----
        #pragma unroll
        for (int i = 0; i < 4; ++i) {
            const int idx = tid + 256 * i;
            const int r = idx >> 4;               // key 0..63
            const int c = (idx & 15) * 4;         // dh
            const float4 t4 = *(const float4*)(V + base + (size_t)(kbase + r) * D_ + c);
            const v4bf p = pack4(t4);
            const int ks = r >> 5, hh = (r >> 3) & 1, j = (r & 7) + 8 * ((r >> 4) & 1);
            VF[ks][c >> 4][hh * 16 + (c & 15) + 0][j] = p[0];
            VF[ks][c >> 4][hh * 16 + (c & 15) + 1][j] = p[1];
            VF[ks][c >> 4][hh * 16 + (c & 15) + 2][j] = p[2];
            VF[ks][c >> 4][hh * 16 + (c & 15) + 3][j] = p[3];
        }
        if (kt < 7) {
            __builtin_prefetch(Km + base + (size_t)(kbase + 64 + (tid >> 4)) * D_ + (tid & 15) * 4, 0, 0);
            __builtin_prefetch(V  + base + (size_t)(kbase + 64 + (tid >> 4)) * D_ + (tid & 15) * 4, 0, 0);
        }
        __syncthreads();

        // S = scale * Q @ K^T   (64x64, K-dim = DH = 64)
        #pragma unroll
        for (int tn = 0; tn < 2; ++tn) {
            v8f accS = {};
            #pragma unroll
            for (int ks = 0; ks < 2; ++ks) {
                const v16bf bb = ld_frag(&KF[ks][(colg >> 4) + tn][lane][0]);
                accS = __builtin_amdgcn_wmma_f32_16x16x32_bf16(
                    false, aq[ks], false, bb, (short)0, accS, false, false);
            }
            #pragma unroll
            for (int r = 0; r < 8; ++r)
                Ss[rowg + 8 * half + r][colg + tn * 16 + l16] = accS[r] * scale;
        }
        __syncthreads();

        // Online softmax per query row; write P in A-fragment order
        if (tid < 64) {
            float mx = -1e30f;
            for (int c = 0; c < 64; ++c) mx = fmaxf(mx, Ss[tid][c]);
            const float mo = m_i[tid];
            const float mn = fmaxf(mo, mx);
            const float cf = __expf(mo - mn);
            float sum = 0.f;
            const int qtl = tid >> 4, rl = tid & 15;
            for (int c = 0; c < 64; ++c) {
                const float p = __expf(Ss[tid][c] - mn);
                sum += p;
                const int kk = c & 31;
                PF[qtl][c >> 5][((kk >> 3) & 1) * 16 + rl][(kk & 7) + 8 * (kk >> 4)] = f2bf(p);
            }
            m_i[tid]  = mn;
            l_i[tid]  = l_i[tid] * cf + sum;
            corr[tid] = cf;
        }
        __syncthreads();

        // Rescale running O, then O += P @ V  (K-dim = 64 keys)
        #pragma unroll
        for (int tn = 0; tn < 2; ++tn)
            #pragma unroll
            for (int r = 0; r < 8; ++r)
                accO[tn][r] *= corr[rowg + 8 * half + r];
        #pragma unroll
        for (int ks = 0; ks < 2; ++ks) {
            const v16bf ap = ld_frag(&PF[mt][ks][lane][0]);
            #pragma unroll
            for (int tn = 0; tn < 2; ++tn) {
                const v16bf bb = ld_frag(&VF[ks][(colg >> 4) + tn][lane][0]);
                accO[tn] = __builtin_amdgcn_wmma_f32_16x16x32_bf16(
                    false, ap, false, bb, (short)0, accO[tn], false, false);
            }
        }
    }
    __syncthreads();

    // Normalize by l_i and write ctx (layout (B,S,H,DH) == (8192, 1024))
    #pragma unroll
    for (int tn = 0; tn < 2; ++tn) {
        #pragma unroll
        for (int r = 0; r < 8; ++r) {
            const int row = rowg + 8 * half + r;
            const float v = accO[tn][r] / l_i[row];
            ctx[base + (size_t)(qbase + row) * D_ + colg + tn * 16 + l16] = v;
        }
    }
}

// ---------------------------------------------------------------------------
// Host orchestration
// ---------------------------------------------------------------------------
extern "C" void kernel_launch(void* const* d_in, const int* in_sizes, int n_in,
                              void* d_out, int out_size, void* d_ws, size_t ws_size,
                              hipStream_t stream) {
    (void)in_sizes; (void)n_in; (void)out_size; (void)ws_size;
    const int*   ids = (const int*)  d_in[0];
    const int*   tts = (const int*)  d_in[1];
    const float* tok = (const float*)d_in[2];
    const float* pos = (const float*)d_in[3];
    const float* typ = (const float*)d_in[4];
    const float* eg  = (const float*)d_in[5];
    const float* eb  = (const float*)d_in[6];
    const float* Wq  = (const float*)d_in[7];
    const float* bq  = (const float*)d_in[8];
    const float* Wk  = (const float*)d_in[9];
    const float* bk  = (const float*)d_in[10];
    const float* Wv  = (const float*)d_in[11];
    const float* bv  = (const float*)d_in[12];
    const float* Wo  = (const float*)d_in[13];
    const float* bo  = (const float*)d_in[14];
    const float* g1  = (const float*)d_in[15];
    const float* bb1 = (const float*)d_in[16];
    const float* g2  = (const float*)d_in[17];
    const float* bb2 = (const float*)d_in[18];
    const float* W1  = (const float*)d_in[19];
    const float* bf1 = (const float*)d_in[20];
    const float* W2  = (const float*)d_in[21];
    const float* bf2 = (const float*)d_in[22];

    float* h  = (float*)d_out;           // running hidden state (B*S, D)
    float* ws = (float*)d_ws;
    const size_t MD = (size_t)MROWS * D_;
    float* x  = ws;                      // LN output
    float* q  = x  + MD;
    float* k  = q  + MD;
    float* v  = k  + MD;
    float* cx = v  + MD;                 // attention context
    float* ff = cx + MD;                 // FFN intermediate (MROWS * FF_)

    const dim3 blk(256);
    const dim3 gD (MROWS / 128, D_  / 128);   // (64, 8)
    const dim3 gFF(MROWS / 128, FF_ / 128);   // (64, 32)

    embed_ln_kernel<<<dim3(MROWS), blk, 0, stream>>>(ids, tts, tok, pos, typ, eg, eb, h);

    for (int l = 0; l < L_; ++l) {
        const size_t oDD = (size_t)l * D_ * D_;
        const size_t oDF = (size_t)l * D_ * FF_;
        const float* wq = Wq + oDD; const float* wk = Wk + oDD;
        const float* wv = Wv + oDD; const float* wo = Wo + oDD;
        const float* w1 = W1 + oDF; const float* w2 = W2 + oDF;

        ln_kernel<<<dim3(MROWS), blk, 0, stream>>>(h, g1 + l * D_, bb1 + l * D_, x);
        gemm_ws<0, 0><<<gD, blk, 0, stream>>>(x, wq, bq + l * D_, nullptr, q, MROWS, D_, D_);
        gemm_ws<0, 0><<<gD, blk, 0, stream>>>(x, wk, bk + l * D_, nullptr, k, MROWS, D_, D_);
        gemm_ws<0, 0><<<gD, blk, 0, stream>>>(x, wv, bv + l * D_, nullptr, v, MROWS, D_, D_);
        attn_kernel<<<dim3(S_ / 64, B_ * H_), blk, 0, stream>>>(q, k, v, cx);
        gemm_ws<0, 1><<<gD, blk, 0, stream>>>(cx, wo, bo + l * D_, h, h, MROWS, D_, D_);
        ln_kernel<<<dim3(MROWS), blk, 0, stream>>>(h, g2 + l * D_, bb2 + l * D_, x);
        gemm_ws<1, 0><<<gFF, blk, 0, stream>>>(x, w1, bf1 + l * FF_, nullptr, ff, MROWS, FF_, D_);
        gemm_ws<0, 1><<<gD, blk, 0, stream>>>(ff, w2, bf2 + l * D_, h, h, MROWS, D_, FF_);
    }
}
